// MultilayerGRU_66640712564742
// MI455X (gfx1250) — matrix operand order repaired and checked by reading, not compiled, not run on hardware
//
#include <hip/hip_runtime.h>
#include <hip/hip_bf16.h>

// ---------------------------------------------------------------------------
// MultilayerGRU on MI455X (gfx1250, wave32, WMMA, double-buffered LDS).
//
// Shapes: B=32, S=512, I=H=O=1024, L=2, 3H=3072.  All GEMMs have K=1024.
//   * bf16 data, fp32 accumulate via v_wmma_f32_16x16x32_bf16.
//   * xproj = inp @ Wx^T + bx  -> big GEMM per layer (M=16384,N=3072)
//   * recurrent scan: two small WMMA kernels per timestep
//   * out = ys @ Why^T + bhy   -> big GEMM (M=16384,N=1024)
// Pipeline per k-block (32): commit regs->LDS(next buf), stage global->regs
// (k+2), compute from LDS(cur buf), one s_barrier.  K templated -> fully
// unrolled, compile-time LDS offsets, partial dependency waits.
// ---------------------------------------------------------------------------

typedef __bf16 v16bf  __attribute__((ext_vector_type(16)));
typedef __bf16 bf16x8 __attribute__((ext_vector_type(8)));
typedef float  v8f    __attribute__((ext_vector_type(8)));

union FragU { v16bf v; bf16x8 h[2]; };

#define B_  32
#define S_  512
#define H_  1024
#define G3_ 3072
#define L_  2

#define LDST 40   // LDS row stride in bf16 elems (32 data + 8 pad = 80 B)

// ---------------------------------------------------------------------------
// Block GEMM core: acc += A[aRow0+BM, :K] @ W[wRow0+BN, :K]^T
// 256 threads = 8 waves (WAVES_M x WAVES_N); wave tile = 16 x 64 (4 accs).
//
// bf16 fragment layouts (ISA 7.12.2, wave32):
//   A 16x32: lane&15 = row M; elems 0..7 -> K=k0+8*half+e,
//            elems 8..15 -> K=k0+16+8*half+(e-8)  => two 16B reads.
//   B 32x16: lane&15 = col N; same K mapping as A => two 16B reads of
//            the row-major [N,K] weight row.
// ---------------------------------------------------------------------------
template<int WAVES_M, int WAVES_N, int KT>
__device__ __forceinline__ void gemm_block_lds(
    const __bf16* __restrict__ A, int lda, int aRow0,
    const __bf16* __restrict__ W, int ldw, int wRow0,
    __bf16* lA, __bf16* lW, v8f acc[4])
{
    constexpr int BM  = WAVES_M * 16;
    constexpr int BN  = WAVES_N * 64;
    constexpr int AT  = BM * 4;                 // 8-elem chunks in A tile (pow2)
    constexpr int WT  = BN * 4;                 // 8-elem chunks in W tile (pow2)
    constexpr int ACH = (AT + 255) / 256;
    constexpr int WCH = (WT + 255) / 256;
    constexpr int NIT = KT / 32;

    const int tid  = threadIdx.x;
    const int wave = tid >> 5, lane = tid & 31;
    const int wm   = wave % WAVES_M, wn = wave / WAVES_M;
    const int half = lane >> 4, mr = lane & 15;

    bf16x8 ra[ACH], rw[WCH];

    // global -> regs for k-block kb (excess threads wrap and duplicate: no divergence)
    auto stage = [&](int kb) {
        const int k0 = kb * 32;
#pragma unroll
        for (int c = 0; c < ACH; ++c) {
            const int idx = (tid + c * 256) & (AT - 1);
            const int row = idx >> 2, cc = (idx & 3) * 8;
            ra[c] = *(const bf16x8*)(A + (size_t)(aRow0 + row) * lda + k0 + cc);
        }
#pragma unroll
        for (int c = 0; c < WCH; ++c) {
            const int idx = (tid + c * 256) & (WT - 1);
            const int row = idx >> 2, cc = (idx & 3) * 8;
            rw[c] = *(const bf16x8*)(W + (size_t)(wRow0 + row) * ldw + k0 + cc);
        }
    };
    // regs -> LDS buffer `buf`
    auto commit = [&](int buf) {
        __bf16* dA = lA + buf * (BM * LDST);
        __bf16* dW = lW + buf * (BN * LDST);
#pragma unroll
        for (int c = 0; c < ACH; ++c) {
            const int idx = (tid + c * 256) & (AT - 1);
            *(bf16x8*)(dA + (idx >> 2) * LDST + (idx & 3) * 8) = ra[c];
        }
#pragma unroll
        for (int c = 0; c < WCH; ++c) {
            const int idx = (tid + c * 256) & (WT - 1);
            *(bf16x8*)(dW + (idx >> 2) * LDST + (idx & 3) * 8) = rw[c];
        }
    };

    stage(0);
    commit(0);
    if (NIT > 1) stage(1);
    __syncthreads();

#pragma unroll
    for (int it = 0; it < NIT; ++it) {
        const int cur = it & 1;
        if (it + 1 < NIT) commit(cur ^ 1);  // idle buffer: last read before prev barrier
        if (it + 2 < NIT) stage(it + 2);

        const __bf16* bufA = lA + cur * (BM * LDST);
        const __bf16* bufW = lW + cur * (BN * LDST);

        // issue all fragment loads, then the 4 WMMAs (partial dscnt waits)
        FragU a, b[4];
        const __bf16* arow = bufA + (wm * 16 + mr) * LDST;
        a.h[0] = *(const bf16x8*)(arow + half * 8);
        a.h[1] = *(const bf16x8*)(arow + 16 + half * 8);
#pragma unroll
        for (int t = 0; t < 4; ++t) {
            const __bf16* wrow = bufW + (wn * 64 + t * 16 + mr) * LDST + half * 8;
            b[t].h[0] = *(const bf16x8*)(wrow);
            b[t].h[1] = *(const bf16x8*)(wrow + 16);
        }
#pragma unroll
        for (int t = 0; t < 4; ++t)
            acc[t] = __builtin_amdgcn_wmma_f32_16x16x32_bf16(
                false, a.v, false, b[t].v, (short)0, acc[t], false, false);

        __syncthreads();
    }
}

__device__ __forceinline__ float sigmoidf_(float x) {
    return 1.0f / (1.0f + __expf(-x));
}

// ---------------------------------------------------------------------------
// Generic GEMM: C[M,N] = A[M,K=1024] @ W[N,K=1024]^T + bias[N]  (fp32 out)
// block 256 thr = 8 waves (4M x 2N) -> 64 x 128 tile; grid (N/128, M/64).
// ---------------------------------------------------------------------------
__global__ void __launch_bounds__(256)
gemm_bf16_bias(const __bf16* __restrict__ A, const __bf16* __restrict__ W,
               const float* __restrict__ bias, float* __restrict__ C,
               int M, int N)
{
    __shared__ __bf16 lA[2 * 64 * LDST];
    __shared__ __bf16 lW[2 * 128 * LDST];
    const int tid = threadIdx.x, wave = tid >> 5, lane = tid & 31;
    const int wm = wave % 4, wn = wave / 4;
    const int m0 = blockIdx.y * 64 + wm * 16;
    const int n0 = blockIdx.x * 128 + wn * 64;

    v8f acc[4] = {};
    gemm_block_lds<4, 2, 1024>(A, 1024, blockIdx.y * 64,
                               W, 1024, blockIdx.x * 128, lA, lW, acc);

    const int half = lane >> 4, nc = lane & 15;
#pragma unroll
    for (int t = 0; t < 4; ++t) {
        const int col = n0 + t * 16 + nc;
        const float bb = bias[col];
#pragma unroll
        for (int j = 0; j < 8; ++j) {
            const int row = m0 + j + 8 * half;
            C[(size_t)row * N + col] = acc[t][j] + bb;
        }
    }
}

// ---------------------------------------------------------------------------
// GRU step part 1: cols 0..2047 of Wh (z|r). pre = h @ W^T + xproj[:,0:2048].
// z stored fp32; r fused into rh = bf16(r*h). grid.x=8, block 256 (2M x 4N).
// ---------------------------------------------------------------------------
__global__ void __launch_bounds__(256)
gru_step_zr(const __bf16* __restrict__ hB, const __bf16* __restrict__ Wh,
            const float* __restrict__ xproj, const float* __restrict__ hF,
            float* __restrict__ zbuf, __bf16* __restrict__ rhB, int s)
{
    __shared__ __bf16 lA[2 * 32 * LDST];
    __shared__ __bf16 lW[2 * 256 * LDST];
    const int tid = threadIdx.x, wave = tid >> 5, lane = tid & 31;
    const int wm = wave % 2, wn = wave / 2;
    const int m0 = wm * 16;
    const int n0 = blockIdx.x * 256 + wn * 64;

    v8f acc[4] = {};
    gemm_block_lds<2, 4, 1024>(hB, H_, 0, Wh, H_, blockIdx.x * 256, lA, lW, acc);

    const int half = lane >> 4, nc = lane & 15;
#pragma unroll
    for (int t = 0; t < 4; ++t) {
        const int col = n0 + t * 16 + nc;          // 0..2047
#pragma unroll
        for (int j = 0; j < 8; ++j) {
            const int b = m0 + j + 8 * half;       // batch row 0..31
            const float pre = acc[t][j] + xproj[((size_t)b * S_ + s) * G3_ + col];
            const float v = sigmoidf_(pre);
            if (col < H_) {
                zbuf[b * H_ + col] = v;            // z gate
            } else {
                const int n = col - H_;            // r gate -> r*h
                rhB[b * H_ + n] = (__bf16)(v * hF[b * H_ + n]);
            }
        }
    }
}

// ---------------------------------------------------------------------------
// GRU step part 2: g = sigmoid(xg + rh @ Whg^T); h = z*h + (1-z)*g.
// Whg = Wh rows 2048..3071 (offset pointer). grid.x=4, block 256.
// ---------------------------------------------------------------------------
__global__ void __launch_bounds__(256)
gru_step_g(const __bf16* __restrict__ rhB, const __bf16* __restrict__ Whg,
           const float* __restrict__ xproj, const float* __restrict__ zbuf,
           float* __restrict__ hF, __bf16* __restrict__ hB,
           __bf16* __restrict__ yout, int s)
{
    __shared__ __bf16 lA[2 * 32 * LDST];
    __shared__ __bf16 lW[2 * 256 * LDST];
    const int tid = threadIdx.x, wave = tid >> 5, lane = tid & 31;
    const int wm = wave % 2, wn = wave / 2;
    const int m0 = wm * 16;
    const int n0 = blockIdx.x * 256 + wn * 64;

    v8f acc[4] = {};
    gemm_block_lds<2, 4, 1024>(rhB, H_, 0, Whg, H_, blockIdx.x * 256, lA, lW, acc);

    const int half = lane >> 4, nc = lane & 15;
#pragma unroll
    for (int t = 0; t < 4; ++t) {
        const int col = n0 + t * 16 + nc;          // 0..1023
#pragma unroll
        for (int j = 0; j < 8; ++j) {
            const int b = m0 + j + 8 * half;
            const float pre = acc[t][j] + xproj[((size_t)b * S_ + s) * G3_ + 2 * H_ + col];
            const float g = sigmoidf_(pre);        // reference uses sigmoid here
            const float z = zbuf[b * H_ + col];
            const float hold = hF[b * H_ + col];
            const float hn = z * hold + (1.0f - z) * g;
            hF[b * H_ + col] = hn;
            hB[b * H_ + col] = (__bf16)hn;
            yout[((size_t)b * S_ + s) * H_ + col] = (__bf16)hn;
        }
    }
}

// ------------------------------ small helpers ------------------------------
__global__ void cast_f32_bf16(const float* __restrict__ src,
                              __bf16* __restrict__ dst, int n)
{
    for (int i = blockIdx.x * blockDim.x + threadIdx.x; i < n;
         i += gridDim.x * blockDim.x)
        dst[i] = (__bf16)src[i];
}

__global__ void init_h_kernel(const float* __restrict__ h0,
                              float* __restrict__ hF, __bf16* __restrict__ hB, int l)
{
    const int i = blockIdx.x * blockDim.x + threadIdx.x;   // 0..B*H-1
    const int b = i >> 10, j = i & (H_ - 1);
    const float v = h0[((size_t)b * L_ + l) * H_ + j];
    hF[i] = v;
    hB[i] = (__bf16)v;
}

__global__ void store_hidden_kernel(const float* __restrict__ hF,
                                    float* __restrict__ outH, int l)
{
    const int i = blockIdx.x * blockDim.x + threadIdx.x;   // 0..B*H-1
    const int b = i >> 10, j = i & (H_ - 1);
    outH[((size_t)b * L_ + l) * H_ + j] = hF[i];
}

// ---------------------------------------------------------------------------
extern "C" void kernel_launch(void* const* d_in, const int* in_sizes, int n_in,
                              void* d_out, int out_size, void* d_ws, size_t ws_size,
                              hipStream_t stream)
{
    (void)in_sizes; (void)n_in; (void)out_size; (void)ws_size;
    const float* x   = (const float*)d_in[0];   // [B,S,I]
    const float* h0  = (const float*)d_in[1];   // [B,L,H]
    const float* Wx  = (const float*)d_in[2];   // [L,3H,I]
    const float* bx  = (const float*)d_in[3];   // [L,3H]
    const float* Wh  = (const float*)d_in[4];   // [L,3H,H]
    const float* Why = (const float*)d_in[5];   // [O,H]
    const float* bhy = (const float*)d_in[6];   // [O]
    float* out       = (float*)d_out;                          // [B,S,O]
    float* hiddenOut = out + (size_t)B_ * S_ * H_;             // [B,L,H]

    // ---- workspace carve-up (256B aligned) ----
    char* ws = (char*)d_ws;
    auto carve = [&](size_t bytes) -> void* {
        void* p = ws; ws += (bytes + 255) & ~(size_t)255; return p;
    };
    const size_t actElems = (size_t)B_ * S_ * H_;              // 16M
    __bf16* act0  = (__bf16*)carve(actElems * 2);
    __bf16* act1  = (__bf16*)carve(actElems * 2);
    __bf16* wxb   = (__bf16*)carve((size_t)G3_ * H_ * 2);
    __bf16* whb   = (__bf16*)carve((size_t)G3_ * H_ * 2);
    __bf16* whyb  = (__bf16*)carve((size_t)H_ * H_ * 2);
    float*  xproj = (float*)carve((size_t)B_ * S_ * G3_ * 4);  // 192 MB
    float*  hF    = (float*)carve((size_t)B_ * H_ * 4);
    __bf16* hB    = (__bf16*)carve((size_t)B_ * H_ * 2);
    float*  zbuf  = (float*)carve((size_t)B_ * H_ * 4);
    __bf16* rhB   = (__bf16*)carve((size_t)B_ * H_ * 2);

    const dim3 blk(256);

    // ---- cast static inputs ----
    cast_f32_bf16<<<dim3((unsigned)((actElems + 255) / 256)), blk, 0, stream>>>(x, act0, (int)actElems);
    cast_f32_bf16<<<dim3((H_ * H_) / 256), blk, 0, stream>>>(Why, whyb, H_ * H_);

    __bf16* actIn = act0;
    __bf16* actOut = act1;

    for (int l = 0; l < L_; ++l) {
        const int wElems = G3_ * H_;
        cast_f32_bf16<<<dim3(wElems / 256), blk, 0, stream>>>(Wx + (size_t)l * wElems, wxb, wElems);
        cast_f32_bf16<<<dim3(wElems / 256), blk, 0, stream>>>(Wh + (size_t)l * wElems, whb, wElems);

        // xproj[b*S+s, :] = inp @ Wx[l]^T + bx[l]   (M=16384, N=3072, K=1024)
        gemm_bf16_bias<<<dim3(G3_ / 128, (B_ * S_) / 64), blk, 0, stream>>>(
            actIn, wxb, bx + (size_t)l * G3_, xproj, B_ * S_, G3_);

        init_h_kernel<<<dim3((B_ * H_) / 256), blk, 0, stream>>>(h0, hF, hB, l);

        for (int s = 0; s < S_; ++s) {
            gru_step_zr<<<dim3(8), blk, 0, stream>>>(hB, whb, xproj, hF, zbuf, rhB, s);
            gru_step_g<<<dim3(4), blk, 0, stream>>>(rhB, whb + (size_t)2 * H_ * H_,
                                                    xproj, zbuf, hF, hB, actOut, s);
        }

        store_hidden_kernel<<<dim3((B_ * H_) / 256), blk, 0, stream>>>(hF, hiddenOut, l);

        __bf16* tmp = actIn; actIn = actOut; actOut = tmp;
    }

    // out = ys @ Why^T + bhy   (M=16384, N=1024, K=1024)
    gemm_bf16_bias<<<dim3(H_ / 128, (B_ * S_) / 64), blk, 0, stream>>>(
        actIn, whyb, bhy, out, B_ * S_, H_);
}
